// SA_Layer_51960514347139
// MI455X (gfx1250) — matrix operand order repaired and checked by reading, not compiled, not run on hardware
//
#include <hip/hip_runtime.h>

#define B_ 4
#define C_ 256
#define N_ 2048
#define EMB_ 256
#define H_ 4
#define DH_ 64
#define BN_EPS 1e-5f

typedef __attribute__((ext_vector_type(16))) _Float16 v16h;
typedef __attribute__((ext_vector_type(8)))  float    v8f;

__device__ __forceinline__ v8f wmma16(v16h a, v16h b, v8f c) {
  // v_wmma_f32_16x16x32_f16
  return __builtin_amdgcn_wmma_f32_16x16x32_f16(false, a, false, b, (short)0, c,
                                                false, false);
}

// ---- WMMA operand loaders (wave32; contiguous k order shared by A and B, not the ISA table order) ----
// All call sites below are arranged so these are contiguous 16+16B per lane.
// 16-bit A (16x32): lane L holds row M=L&15; halves index K = kb+i, kb = 0
// (lanes 0-15) or 16 (lanes 16-31).  Source row-major (row stride in halves).
__device__ __forceinline__ v16h load_A_rm(const _Float16* p, int stride, int lane) {
  const int m  = lane & 15;
  const int kb = (lane & 16) ? 16 : 0;
  const _Float16* r = p + (size_t)m * stride + kb;
  v16h a;
#pragma unroll
  for (int i = 0; i < 16; ++i) a[i] = r[i];
  return a;
}
// 16-bit B (32x16): lane L holds column N=L&15; halves index K = kb+i, kb = 0
// (lanes 0-15) or 16 (lanes 16-31).  Source stored column-major: element(k, n)
// at p[n*stride + k]  ->  contiguous 32B per lane.
__device__ __forceinline__ v16h load_B_cm(const _Float16* p, int stride, int lane) {
  const int n  = lane & 15;
  const int kb = (lane & 16) ? 16 : 0;
  const _Float16* r = p + (size_t)n * stride + kb;
  v16h b;
#pragma unroll
  for (int i = 0; i < 16; ++i) b[i] = r[i];
  return b;
}
// f32 C/D (16x16): VGPR r -> row M = r + 8*(lane>=16), column N = lane&15.

// ---------------- workspace layout ----------------
static constexpr size_t SZ_CN   = (size_t)B_ * C_ * N_;        // 2,097,152
static constexpr size_t ACC_OFF = 0;                           // 514 f32
static constexpr size_t XIT_OFF = 4096;                        // f16 xi^T (B,N,C)
static constexpr size_t PTS_OFF = XIT_OFF + SZ_CN * 2;         // f16 xyz^T (B,N,C)
static constexpr size_t WH_OFF  = PTS_OFF + SZ_CN * 2;         // f16 Wq,Wk,Wv,Wt
static constexpr size_t SQ_OFF  = WH_OFF + (size_t)4 * 65536 * 2;
static constexpr size_t QH_OFF  = SQ_OFF + (size_t)B_ * N_ * 4; // f16 (B,H,N,DH)
static constexpr size_t KH_OFF  = QH_OFF + SZ_CN * 2;          // f16 (B,H,N,DH)
static constexpr size_t VT_OFF  = KH_OFF + SZ_CN * 2;          // f16 v^T (B,H,DH,N)
static constexpr size_t TH_OFF  = VT_OFF + SZ_CN * 2;          // f16 (xi-x_r)^T (B,N,C)
static constexpr size_t DM_OFF  = TH_OFF + SZ_CN * 2;          // f16 Dmat (B,N,N)
static constexpr size_t Y_OFF   = DM_OFF + (size_t)B_ * N_ * N_ * 2; // f32 y (B,C,N)

// ---------------- kernels ----------------
__global__ void prep_data(const float* __restrict__ x, const float* __restrict__ xyz,
                          _Float16* __restrict__ xiT, _Float16* __restrict__ ptsT,
                          float* __restrict__ sq) {
  int i = blockIdx.x * blockDim.x + threadIdx.x;  // over B*N
  if (i >= B_ * N_) return;
  int b = i / N_, n = i % N_;
  const float* xp = x   + (size_t)b * C_ * N_ + n;
  const float* zp = xyz + (size_t)b * C_ * N_ + n;
  _Float16* xo = xiT  + (size_t)i * C_;
  _Float16* po = ptsT + (size_t)i * C_;
  float acc = 0.f;
  for (int c = 0; c < C_; ++c) {
    float xv = xp[(size_t)c * N_], zv = zp[(size_t)c * N_];
    xo[c] = (_Float16)(xv + zv);
    po[c] = (_Float16)zv;
    acc += zv * zv;
  }
  sq[i] = acc;
}

__global__ void prep_w(const float* __restrict__ Wq, const float* __restrict__ Wk,
                       const float* __restrict__ Wv, const float* __restrict__ Wt,
                       _Float16* __restrict__ o) {
  int i = blockIdx.x * blockDim.x + threadIdx.x;
  if (i >= 4 * 65536) return;
  int w = i >> 16, j = i & 65535;
  const float* s = (w == 0) ? Wq : (w == 1) ? Wk : (w == 2) ? Wv : Wt;
  o[i] = (_Float16)s[j];
}

__global__ __launch_bounds__(32) void proj_qkv(const _Float16* __restrict__ xiT,
                                               const _Float16* __restrict__ wh,
                                               _Float16* __restrict__ qh,
                                               _Float16* __restrict__ kh,
                                               _Float16* __restrict__ vT) {
  const int lane = threadIdx.x;
  int bid = blockIdx.x;                 // B * (N/16) * (EMB/16)
  const int et = bid & 15;  bid >>= 4;
  const int nt = bid & 127; bid >>= 7;
  const int b  = bid;
  const int n0 = nt * 16, e0 = et * 16;
  const _Float16* xb = xiT + ((size_t)b * N_ + n0) * C_;
  v8f cq = {}, ck = {}, cv = {};
  for (int kc = 0; kc < C_; kc += 32) {
    v16h a  = load_A_rm(xb + kc, C_, lane);                        // xi^T tile
    v16h bq = load_B_cm(wh + 0 * 65536 + (size_t)e0 * C_ + kc, C_, lane);
    v16h bk = load_B_cm(wh + 1 * 65536 + (size_t)e0 * C_ + kc, C_, lane);
    v16h bv = load_B_cm(wh + 2 * 65536 + (size_t)e0 * C_ + kc, C_, lane);
    cq = wmma16(a, bq, cq);
    ck = wmma16(a, bk, ck);
    cv = wmma16(a, bv, cv);
  }
  const int col = lane & 15, hi = (lane >> 4) & 1;
  const int e = e0 + col, h = e >> 6, d = e & 63;
#pragma unroll
  for (int r = 0; r < 8; ++r) {
    int n = n0 + r + 8 * hi;
    size_t idx = (((size_t)b * H_ + h) * N_ + n) * DH_ + d;
    qh[idx] = (_Float16)cq[r];
    kh[idx] = (_Float16)ck[r];
    // v stored transposed: (B,H,DH,N) so p@v B-tiles are contiguous per lane
    vT[(((size_t)b * H_ + h) * DH_ + d) * N_ + n] = (_Float16)cv[r];
  }
}

__global__ __launch_bounds__(32) void dmat_k(const _Float16* __restrict__ ptsT,
                                             const float* __restrict__ sq,
                                             _Float16* __restrict__ dm) {
  const int lane = threadIdx.x;
  int bid = blockIdx.x;                 // B * (N/16) * (N/16)
  const int mt = bid & 127; bid >>= 7;
  const int nt = bid & 127; bid >>= 7;
  const int b  = bid;
  const int n0 = nt * 16, m0 = mt * 16;
  const _Float16* pb = ptsT + (size_t)b * N_ * C_;
  v8f g = {};
  for (int kc = 0; kc < C_; kc += 32) {
    v16h a  = load_A_rm(pb + (size_t)n0 * C_ + kc, C_, lane);  // pts rows (n,k=c)
    v16h bm = load_B_cm(pb + (size_t)m0 * C_ + kc, C_, lane);  // pts^T cols (k=c,m)
    g = wmma16(a, bm, g);
  }
  const int col = lane & 15, hi = (lane >> 4) & 1;
  const float sqm = sq[b * N_ + m0 + col];
#pragma unroll
  for (int r = 0; r < 8; ++r) {
    int n = n0 + r + 8 * hi;
    float d2 = sq[b * N_ + n] + sqm - 2.f * g[r];
    dm[(size_t)b * N_ * N_ + (size_t)n * N_ + m0 + col] =
        (_Float16)sqrtf(fmaxf(d2, 0.f));
  }
}

__global__ __launch_bounds__(32) void attn_k(const _Float16* __restrict__ qh,
                                             const _Float16* __restrict__ kh,
                                             const _Float16* __restrict__ vT,
                                             const _Float16* __restrict__ dm,
                                             const _Float16* __restrict__ xiT,
                                             _Float16* __restrict__ th,
                                             float* __restrict__ accum) {
  __shared__ _Float16 lds[16 * 32];     // exp(e-M) tile, C/D -> A re-layout
  const int lane = threadIdx.x;
  int bid = blockIdx.x;                 // B * H * (N/16)
  const int nt = bid & 127; bid >>= 7;
  const int h  = bid & 3;   bid >>= 2;
  const int b  = bid;
  const int n0 = nt * 16;
  const size_t headoff = ((size_t)b * H_ + h) * N_ * DH_;
  const _Float16* qptr = qh + headoff;
  const _Float16* kptr = kh + headoff;
  const _Float16* vptr = vT + headoff;  // (DH, N) within head
  const _Float16* dmb  = dm + (size_t)b * N_ * N_;
  const int col = lane & 15, hi = (lane >> 4) & 1;

  // Hoist the two A tiles of q (K = 0..31, 32..63)
  v16h aq0 = load_A_rm(qptr + (size_t)n0 * DH_ + 0,  DH_, lane);
  v16h aq1 = load_A_rm(qptr + (size_t)n0 * DH_ + 32, DH_, lane);

  float M[8], Z[8], S1[8], SD[8];
#pragma unroll
  for (int r = 0; r < 8; ++r) { M[r] = -1e30f; Z[r] = 0.f; S1[r] = 0.f; SD[r] = 0.f; }
  v8f oacc[4];
  {
    v8f z = {};
#pragma unroll
    for (int t = 0; t < 4; ++t) oacc[t] = z;
  }

  for (int m0 = 0; m0 < N_; m0 += 32) {
    if (m0 + 32 < N_) {                 // -> global_prefetch_b8
      __builtin_prefetch(kptr + (size_t)(m0 + 32) * DH_, 0, 0);
      __builtin_prefetch(vptr + (size_t)(m0 + 32), 0, 0);
    }
    // ---- energy: q (16x64) x k^T (64x32) -> two 16x16 tiles
    // k stored (m, d): B[k=d][col=m] via column-major loader, contiguous in d.
    v8f e0 = {}, e1 = {};
    {
      v16h b0 = load_B_cm(kptr + (size_t)(m0)      * DH_ + 0,  DH_, lane);
      v16h b1 = load_B_cm(kptr + (size_t)(m0)      * DH_ + 32, DH_, lane);
      e0 = wmma16(aq0, b0, e0);
      e0 = wmma16(aq1, b1, e0);
      v16h b2 = load_B_cm(kptr + (size_t)(m0 + 16) * DH_ + 0,  DH_, lane);
      v16h b3 = load_B_cm(kptr + (size_t)(m0 + 16) * DH_ + 32, DH_, lane);
      e1 = wmma16(aq0, b2, e1);
      e1 = wmma16(aq1, b3, e1);
    }
    // ---- chunk row max (reduce across the 16 columns of each half)
    float mloc[8];
#pragma unroll
    for (int r = 0; r < 8; ++r) mloc[r] = fmaxf(e0[r], e1[r]);
    for (int mask = 1; mask < 16; mask <<= 1) {
#pragma unroll
      for (int r = 0; r < 8; ++r)
        mloc[r] = fmaxf(mloc[r], __shfl_xor(mloc[r], mask, 32));
    }
    // ---- online rescale
    float alpha[8];
#pragma unroll
    for (int r = 0; r < 8; ++r) {
      float Mn = fmaxf(M[r], mloc[r]);
      float a  = __expf(M[r] - Mn);
      S1[r] = a * (S1[r] + (M[r] - Mn) * Z[r]);
      Z[r] *= a;
      SD[r] *= a;
      alpha[r] = a;
      M[r] = Mn;
    }
#pragma unroll
    for (int t = 0; t < 4; ++t) {
#pragma unroll
      for (int r = 0; r < 8; ++r) oacc[t][r] *= alpha[r];
    }
    // ---- p = exp(e-M); accumulate Z, entropy, distance; stash p to LDS
    float zacc[8], s1acc[8], sdacc[8];
#pragma unroll
    for (int r = 0; r < 8; ++r) { zacc[r] = 0.f; s1acc[r] = 0.f; sdacc[r] = 0.f; }
#pragma unroll
    for (int ct = 0; ct < 2; ++ct) {
#pragma unroll
      for (int r = 0; r < 8; ++r) {
        float ev = (ct == 0) ? e0[r] : e1[r];
        int n = n0 + r + 8 * hi;
        int m = m0 + ct * 16 + col;
        float t = ev - M[r];
        float p = __expf(t);
        float D = (float)dmb[(size_t)n * N_ + m];
        zacc[r]  += p;
        s1acc[r] += t * p;
        sdacc[r] += p * D;
        lds[(r + 8 * hi) * 32 + ct * 16 + col] = (_Float16)p;
      }
    }
    for (int mask = 1; mask < 16; mask <<= 1) {
#pragma unroll
      for (int r = 0; r < 8; ++r) {
        zacc[r]  += __shfl_xor(zacc[r],  mask, 32);
        s1acc[r] += __shfl_xor(s1acc[r], mask, 32);
        sdacc[r] += __shfl_xor(sdacc[r], mask, 32);
      }
    }
#pragma unroll
    for (int r = 0; r < 8; ++r) { Z[r] += zacc[r]; S1[r] += s1acc[r]; SD[r] += sdacc[r]; }
    __syncthreads();                    // single-wave block: orders LDS store->load
    // ---- O += p (16x32) x v (32x64); v^T (d, m): contiguous B-tiles
    v16h ap = load_A_rm(lds, 32, lane);
#pragma unroll
    for (int t = 0; t < 4; ++t) {
      v16h bv = load_B_cm(vptr + (size_t)(t * 16) * N_ + m0, N_, lane);
      oacc[t] = wmma16(ap, bv, oacc[t]);
    }
    __syncthreads();
  }

  // ---- per-row scalars
  if (col == 0) {
    float ds = 0.f, es = 0.f;
#pragma unroll
    for (int r = 0; r < 8; ++r) {
      ds += SD[r] / Z[r];
      es += __logf(Z[r]) - S1[r] / Z[r];
    }
    const float scale = 1.0f / (float)(B_ * H_ * N_);
    atomicAdd(&accum[0], ds * scale);
    atomicAdd(&accum[1], es * scale);
  }
  // ---- th = f16(xi - x_r), stored (B,N,C); x_r = O / Z
  const _Float16* xb2 = xiT + (size_t)b * N_ * C_;
#pragma unroll
  for (int t = 0; t < 4; ++t) {
    int c = h * 64 + t * 16 + col;
#pragma unroll
    for (int r = 0; r < 8; ++r) {
      int n = n0 + r + 8 * hi;
      size_t idx = (size_t)n * C_ + c;
      float xr = oacc[t][r] / Z[r];
      float xi = (float)xb2[idx];
      th[(size_t)b * N_ * C_ + idx] = (_Float16)(xi - xr);
    }
  }
}

__global__ __launch_bounds__(32) void trans_k(const _Float16* __restrict__ wth,
                                              const _Float16* __restrict__ th,
                                              const float* __restrict__ bt,
                                              float* __restrict__ y,
                                              float* __restrict__ accum) {
  const int lane = threadIdx.x;
  int bid = blockIdx.x;                 // B * (N/16) * (C/16)
  const int ot = bid & 15;  bid >>= 4;
  const int nt = bid & 127; bid >>= 7;
  const int b  = bid;
  const int o0 = ot * 16, n0 = nt * 16;
  const _Float16* tb = th + ((size_t)b * N_ + n0) * C_;  // (N, C) rows
  v8f c = {};
  for (int kc = 0; kc < C_; kc += 32) {
    v16h a  = load_A_rm(wth + (size_t)o0 * C_ + kc, C_, lane);   // Wt rows (o, c)
    v16h bb = load_B_cm(tb + kc, C_, lane);                      // t^T: (k=c, n)
    c = wmma16(a, bb, c);
  }
  const int col = lane & 15, hi = (lane >> 4) & 1;
  float ys[8], y2[8];
#pragma unroll
  for (int r = 0; r < 8; ++r) {
    int o = o0 + r + 8 * hi;
    float yv = c[r] + bt[o];
    y[((size_t)b * C_ + o) * N_ + n0 + col] = yv;
    ys[r] = yv;
    y2[r] = yv * yv;
  }
  for (int mask = 1; mask < 16; mask <<= 1) {
#pragma unroll
    for (int r = 0; r < 8; ++r) {
      ys[r] += __shfl_xor(ys[r], mask, 32);
      y2[r] += __shfl_xor(y2[r], mask, 32);
    }
  }
  if (col == 0) {
#pragma unroll
    for (int r = 0; r < 8; ++r) {
      int o = o0 + r + 8 * hi;
      atomicAdd(&accum[2 + o], ys[r]);
      atomicAdd(&accum[2 + 256 + o], y2[r]);
    }
  }
}

__global__ void finalize_k(const float* __restrict__ x, const float* __restrict__ xyz,
                           const float* __restrict__ y,
                           const float* __restrict__ gamma,
                           const float* __restrict__ beta,
                           const float* __restrict__ accum,
                           float* __restrict__ out) {
  size_t i = (size_t)blockIdx.x * blockDim.x + threadIdx.x;
  const size_t total = SZ_CN;
  if (i < total) {
    int c = (int)((i / N_) % C_);
    const float invBN = 1.0f / (float)(B_ * N_);
    float mean = accum[2 + c] * invBN;
    float var  = accum[2 + 256 + c] * invBN - mean * mean;
    float yhat = (y[i] - mean) * rsqrtf(var + BN_EPS);
    float v = gamma[c] * yhat + beta[c];
    v = fmaxf(v, 0.f);
    out[i] = (x[i] + xyz[i]) + v;
  }
  if (i == 0) {
    out[total]     = accum[0];  // attn_distance
    out[total + 1] = accum[1];  // attn_entropy
  }
}

// ---------------- launch ----------------
extern "C" void kernel_launch(void* const* d_in, const int* in_sizes, int n_in,
                              void* d_out, int out_size, void* d_ws, size_t ws_size,
                              hipStream_t stream) {
  const float* x     = (const float*)d_in[0];
  const float* xyz   = (const float*)d_in[1];
  const float* Wq    = (const float*)d_in[2];
  const float* Wk    = (const float*)d_in[3];
  const float* Wv    = (const float*)d_in[4];
  const float* Wt    = (const float*)d_in[5];
  const float* bt    = (const float*)d_in[6];
  const float* gamma = (const float*)d_in[7];
  const float* beta  = (const float*)d_in[8];

  char* ws = (char*)d_ws;
  float*    accum = (float*)(ws + ACC_OFF);
  _Float16* xiT   = (_Float16*)(ws + XIT_OFF);
  _Float16* ptsT  = (_Float16*)(ws + PTS_OFF);
  _Float16* wh    = (_Float16*)(ws + WH_OFF);
  float*    sq    = (float*)(ws + SQ_OFF);
  _Float16* qh    = (_Float16*)(ws + QH_OFF);
  _Float16* kh    = (_Float16*)(ws + KH_OFF);
  _Float16* vT    = (_Float16*)(ws + VT_OFF);
  _Float16* th    = (_Float16*)(ws + TH_OFF);
  _Float16* dmw   = (_Float16*)(ws + DM_OFF);
  float*    yw    = (float*)(ws + Y_OFF);

  hipMemsetAsync(accum, 0, 514 * sizeof(float), stream);

  prep_data<<<(B_ * N_ + 255) / 256, 256, 0, stream>>>(x, xyz, xiT, ptsT, sq);
  prep_w<<<(4 * 65536 + 255) / 256, 256, 0, stream>>>(Wq, Wk, Wv, Wt, wh);
  proj_qkv<<<B_ * (N_ / 16) * (EMB_ / 16), 32, 0, stream>>>(xiT, wh, qh, kh, vT);
  dmat_k<<<B_ * (N_ / 16) * (N_ / 16), 32, 0, stream>>>(ptsT, sq, dmw);
  attn_k<<<B_ * H_ * (N_ / 16), 32, 0, stream>>>(qh, kh, vT, dmw, xiT, th, accum);
  trans_k<<<B_ * (N_ / 16) * (C_ / 16), 32, 0, stream>>>(wh + 3 * 65536, th, bt, yw, accum);
  finalize_k<<<(int)((SZ_CN + 255) / 256), 256, 0, stream>>>(x, xyz, yw, gamma, beta,
                                                             accum, (float*)d_out);
}
